// GNNEncoder_38551626449391
// MI455X (gfx1250) — matrix operand order repaired and checked by reading, not compiled, not run on hardware
//
#include <hip/hip_runtime.h>
#include <cstdint>
#include <cstddef>

// ---------------- problem constants (match reference) ----------------
#define NNODES 200000
#define NEDGES 800000
#define NGRAPH 8000
#define FIN    64
#define NHEADS 4
#define CHAN   64
#define HCDIM  (NHEADS * CHAN)     // 256
#define ETOT   (NEDGES + NNODES)   // 1,000,000 edges incl. self-loops
#define NEG_SLOPE 0.2f

typedef __attribute__((ext_vector_type(16))) __bf16 v16bf;
typedef __attribute__((ext_vector_type(8)))  float  v8f;

// ---------------- helpers ----------------
__device__ __forceinline__ unsigned short f2bf_u(float f) {
  unsigned u = __float_as_uint(f);
  unsigned r = u + 0x7FFFu + ((u >> 16) & 1u);   // round-to-nearest-even
  return (unsigned short)(r >> 16);
}

// order-preserving float <-> uint mapping (for atomic max on floats)
__device__ __forceinline__ unsigned fenc(float f) {
  unsigned u = __float_as_uint(f);
  return (u & 0x80000000u) ? ~u : (u | 0x80000000u);
}
__device__ __forceinline__ float fdec(unsigned e) {
  return (e & 0x80000000u) ? __uint_as_float(e ^ 0x80000000u)
                           : __uint_as_float(~e);
}

__device__ __forceinline__ void atomicAddF(float* p, float v) {
  // lowers to no-return global_atomic_add_f32 at device scope
  __hip_atomic_fetch_add(p, v, __ATOMIC_RELAXED, __HIP_MEMORY_SCOPE_AGENT);
}

__device__ __forceinline__ void edge_nodes(int e, const int* __restrict__ src,
                                           const int* __restrict__ dst,
                                           int& s_, int& d_) {
  if (e < NEDGES) { s_ = src[e]; d_ = dst[e]; }
  else            { s_ = e - NEDGES; d_ = s_; }   // self loops appended
}

// ---------------- f32 -> bf16 conversion pass (4 elems / thread) ----------------
__global__ void cvt_f32_bf16(const float* __restrict__ in,
                             unsigned short* __restrict__ out, size_t n4) {
  size_t i = (size_t)blockIdx.x * blockDim.x + threadIdx.x;
  if (i >= n4) return;
  float4 f = reinterpret_cast<const float4*>(in)[i];
  uint2 u;
  u.x = (unsigned)f2bf_u(f.x) | ((unsigned)f2bf_u(f.y) << 16);
  u.y = (unsigned)f2bf_u(f.z) | ((unsigned)f2bf_u(f.w) << 16);
  reinterpret_cast<uint2*>(out)[i] = u;
}

// ---------------- WMMA GEMM: out[M,Nout] = X[M,K] @ W[K,Nout] (bf16 in memory, f32 acc)
// grid: (M/64, Nout/64), block: 128 threads = 4 waves.
// Each wave computes a 16x64 strip: one A fragment feeds 4 WMMA tiles per k-step.
// K is a template parameter (64 or 256) so the k-loop fully unrolls.
// A-frag per lane = two contiguous 16B chunks; B-frag per lane = 16 scalar bf16 loads strided by Nout.
template <int K>
__global__ __launch_bounds__(128)
void gemm_bf16_wmma(const __bf16* __restrict__ X, const __bf16* __restrict__ W,
                    float* __restrict__ out, int Nout) {
  const int lane    = threadIdx.x & 31;
  const int wave    = threadIdx.x >> 5;
  const int rowBase = (blockIdx.x * 4 + wave) * 16;
  const int colBase = blockIdx.y * 64;
  const int hi      = lane >> 4;     // 0 | 1 (K-half select for A)
  const int lm      = lane & 15;

  v8f acc0 = {}, acc1 = {}, acc2 = {}, acc3 = {};
  const __bf16* xrow = X + (size_t)(rowBase + lm) * K;

#pragma unroll
  for (int k0 = 0; k0 < K; k0 += 32) {
    // A fragment (ISA 16-bit 16x32 layout): lane holds row M=lm,
    // halfs 0..7  = K[k0 + hi*8 .. +7], halfs 8..15 = K[k0 + 16 + hi*8 .. +7]
    v16bf a;
    reinterpret_cast<uint4*>(&a)[0] =
        *reinterpret_cast<const uint4*>(xrow + k0 + hi * 8);
    reinterpret_cast<uint4*>(&a)[1] =
        *reinterpret_cast<const uint4*>(xrow + k0 + 16 + hi * 8);

    // B fragments: lane holds column N=lm of each tile, K in the same order as A
    const __bf16* wcol = W + (size_t)k0 * Nout + colBase + lm;
    v16bf b0, b1, b2, b3;
#pragma unroll
    for (int s = 0; s < 16; ++s) {
      const __bf16* wp = wcol + (size_t)((s & 7) + hi * 8 + (s >> 3) * 16) * Nout;
      b0[s] = wp[0];
      b1[s] = wp[16];
      b2[s] = wp[32];
      b3[s] = wp[48];
    }

    acc0 = __builtin_amdgcn_wmma_f32_16x16x32_bf16(false, a, false, b0, (short)0, acc0, false, false);
    acc1 = __builtin_amdgcn_wmma_f32_16x16x32_bf16(false, a, false, b1, (short)0, acc1, false, false);
    acc2 = __builtin_amdgcn_wmma_f32_16x16x32_bf16(false, a, false, b2, (short)0, acc2, false, false);
    acc3 = __builtin_amdgcn_wmma_f32_16x16x32_bf16(false, a, false, b3, (short)0, acc3, false, false);
  }

  // D layout: VGPR i -> row (i + 8*hi), col lm
  float* obase = out + (size_t)(rowBase + hi * 8) * Nout + colBase + lm;
#pragma unroll
  for (int i = 0; i < 8; ++i) {
    float* orow = obase + (size_t)i * Nout;
    orow[0]  = acc0[i];
    orow[16] = acc1[i];
    orow[32] = acc2[i];
    orow[48] = acc3[i];
  }
}

// ---------------- attention source/dest scores ----------------
// ai[n,h] = sum_c h[n,h,c] * a_src[h,c] ; aj likewise with a_dst
__global__ void att_scores(const float* __restrict__ h,
                           const float* __restrict__ a_src,
                           const float* __restrict__ a_dst,
                           float* __restrict__ ai, float* __restrict__ aj,
                           int Hh) {
  int tid = blockIdx.x * blockDim.x + threadIdx.x;
  if (tid >= NNODES * Hh) return;
  int hd = tid % Hh;
  const float* hp = h + (size_t)tid * CHAN;   // [N,H,C] contiguous
  const float* as = a_src + hd * CHAN;
  const float* ad = a_dst + hd * CHAN;
  float si = 0.f, sj = 0.f;
#pragma unroll 8
  for (int c = 0; c < CHAN; ++c) {
    float v = hp[c];
    si += v * as[c];
    sj += v * ad[c];
  }
  ai[tid] = si;
  aj[tid] = sj;
}

// ---------------- generic fill ----------------
__global__ void fill_u32(unsigned* __restrict__ p, size_t n, unsigned val) {
  size_t i = (size_t)blockIdx.x * blockDim.x + threadIdx.x;
  if (i < n) p[i] = val;
}

// ---------------- edge pass 1: segment max of leaky_relu scores over dst ----------------
__global__ void edge_max(const int* __restrict__ src, const int* __restrict__ dst,
                         const float* __restrict__ ai, const float* __restrict__ aj,
                         unsigned* __restrict__ mu, int Hh) {
  int tid = blockIdx.x * blockDim.x + threadIdx.x;
  if (tid >= ETOT * Hh) return;
  int hd = tid % Hh, e = tid / Hh;
  int s_, d_;
  edge_nodes(e, src, dst, s_, d_);
  float v = ai[s_ * Hh + hd] + aj[d_ * Hh + hd];
  v = v > 0.f ? v : NEG_SLOPE * v;
  atomicMax(&mu[d_ * Hh + hd], fenc(v));
}

// ---------------- edge pass 2: p = exp(e - m[dst]); segment sum over dst ----------------
__global__ void edge_p(const int* __restrict__ src, const int* __restrict__ dst,
                       const float* __restrict__ ai, const float* __restrict__ aj,
                       const unsigned* __restrict__ mu, float* __restrict__ sden,
                       float* __restrict__ pbuf, int Hh) {
  int tid = blockIdx.x * blockDim.x + threadIdx.x;
  if (tid >= ETOT * Hh) return;
  int hd = tid % Hh, e = tid / Hh;
  int s_, d_;
  edge_nodes(e, src, dst, s_, d_);
  float v = ai[s_ * Hh + hd] + aj[d_ * Hh + hd];
  v = v > 0.f ? v : NEG_SLOPE * v;
  float mm = fdec(mu[d_ * Hh + hd]);
  float pv = __expf(v - mm);
  pbuf[tid] = pv;
  atomicAddF(&sden[d_ * Hh + hd], pv);
}

// ---------------- edge pass 3: out[dst,h,:] += alpha * h[src,h,:] ----------------
// one thread per (edge, head, 4-channel chunk)
__global__ void edge_agg(const int* __restrict__ src, const int* __restrict__ dst,
                         const float* __restrict__ h, const float* __restrict__ pbuf,
                         const float* __restrict__ sden, float* __restrict__ out,
                         int Hh) {
  size_t tid = (size_t)blockIdx.x * blockDim.x + threadIdx.x;
  size_t total = (size_t)ETOT * Hh * (CHAN / 4);
  if (tid >= total) return;
  int    c4 = (int)(tid & (CHAN / 4 - 1));   // 16 chunks of float4
  size_t t2 = tid >> 4;
  int    hd = (int)(t2 % Hh);
  int    e  = (int)(t2 / Hh);
  int s_, d_;
  edge_nodes(e, src, dst, s_, d_);
  float alpha = pbuf[(size_t)e * Hh + hd] / (sden[d_ * Hh + hd] + 1e-16f);
  const float4 v =
      *reinterpret_cast<const float4*>(h + ((size_t)s_ * Hh + hd) * CHAN + c4 * 4);
  float* op = out + ((size_t)d_ * Hh + hd) * CHAN + c4 * 4;
  atomicAddF(op + 0, v.x * alpha);
  atomicAddF(op + 1, v.y * alpha);
  atomicAddF(op + 2, v.z * alpha);
  atomicAddF(op + 3, v.w * alpha);
}

// ---------------- bias + ELU, f32 in -> bf16 out (feeds next layer's GEMM) -------
__global__ void bias_elu_bf(const float* __restrict__ x, const float* __restrict__ b,
                            unsigned short* __restrict__ out, size_t n, int mask) {
  size_t i = (size_t)blockIdx.x * blockDim.x + threadIdx.x;
  if (i >= n) return;
  float v = x[i] + b[(int)i & mask];   // Nout is a power of two
  v = v > 0.f ? v : (__expf(v) - 1.f);
  out[i] = f2bf_u(v);
}

// ---------------- bias only, f32 in place (final layer, feeds pooling) ----------
__global__ void bias_f32(float* __restrict__ x, const float* __restrict__ b,
                         size_t n, int mask) {
  size_t i = (size_t)blockIdx.x * blockDim.x + threadIdx.x;
  if (i >= n) return;
  x[i] = x[i] + b[(int)i & mask];
}

// ---------------- per-graph max pool ----------------
__global__ void pool_max(const float* __restrict__ h2, const int* __restrict__ batch,
                         unsigned* __restrict__ pu) {
  int tid = blockIdx.x * blockDim.x + threadIdx.x;
  if (tid >= NNODES * CHAN) return;
  int c = tid & (CHAN - 1);
  int n = tid >> 6;
  atomicMax(&pu[(size_t)batch[n] * CHAN + c], fenc(h2[tid]));
}
__global__ void pool_dec(const unsigned* __restrict__ pu, float* __restrict__ out) {
  int tid = blockIdx.x * blockDim.x + threadIdx.x;
  if (tid >= NGRAPH * CHAN) return;
  unsigned e = pu[tid];
  out[tid] = (e == 0u) ? -__builtin_inff() : fdec(e);
}

// ---------------- launch ----------------
extern "C" void kernel_launch(void* const* d_in, const int* in_sizes, int n_in,
                              void* d_out, int out_size, void* d_ws, size_t ws_size,
                              hipStream_t stream) {
  (void)in_sizes; (void)n_in; (void)out_size; (void)ws_size;
  const float* x     = (const float*)d_in[0];
  const int*   eidx  = (const int*)d_in[1];
  const int*   batch = (const int*)d_in[2];
  const float* W0  = (const float*)d_in[3];
  const float* as0 = (const float*)d_in[4];
  const float* ad0 = (const float*)d_in[5];
  const float* b0  = (const float*)d_in[6];
  const float* W1  = (const float*)d_in[7];
  const float* as1 = (const float*)d_in[8];
  const float* ad1 = (const float*)d_in[9];
  const float* b1  = (const float*)d_in[10];
  const float* W2  = (const float*)d_in[11];
  const float* as2 = (const float*)d_in[12];
  const float* ad2 = (const float*)d_in[13];
  const float* b2  = (const float*)d_in[14];

  const int* src = eidx;            // edge_index[0]
  const int* dst = eidx + NEDGES;   // edge_index[1]

  // workspace layout (bytes)
  char* ws = (char*)d_ws;
  size_t off = 0;
  auto alloc = [&](size_t bytes) {
    void* p = ws + off;
    off += (bytes + 255) & ~(size_t)255;   // keep 256B alignment
    return p;
  };
  float*          bufA = (float*)alloc((size_t)NNODES * HCDIM * 4);  // h = x@W (f32)
  float*          bufB = (float*)alloc((size_t)NNODES * HCDIM * 4);  // aggregation out (f32)
  unsigned short* xbf  = (unsigned short*)alloc((size_t)NNODES * HCDIM * 2); // bf16 GEMM input
  unsigned short* wbf  = (unsigned short*)alloc((size_t)HCDIM * HCDIM * 2);  // bf16 weights
  float*          ai   = (float*)alloc((size_t)NNODES * NHEADS * 4);
  float*          aj   = (float*)alloc((size_t)NNODES * NHEADS * 4);
  unsigned*       mu   = (unsigned*)alloc((size_t)NNODES * NHEADS * 4);
  float*          sden = (float*)alloc((size_t)NNODES * NHEADS * 4);
  float*          pbuf = (float*)alloc((size_t)ETOT * NHEADS * 4);
  unsigned*       pu   = (unsigned*)alloc((size_t)NGRAPH * CHAN * 4);

  auto fill0 = [&](void* p, size_t n) {
    fill_u32<<<(unsigned)((n + 255) / 256), 256, 0, stream>>>((unsigned*)p, n, 0u);
  };
  auto cvt = [&](const float* in, unsigned short* out, size_t n) {
    size_t n4 = n / 4;
    cvt_f32_bf16<<<(unsigned)((n4 + 255) / 256), 256, 0, stream>>>(in, out, n4);
  };

  auto layer = [&](int K, int Hh, const float* W, const float* asv,
                   const float* adv, const float* bias, int isLast) {
    const int Nout = Hh * CHAN;
    // 0) weights -> bf16
    cvt(W, wbf, (size_t)K * Nout);
    // 1) projection via WMMA: bufA = xbf @ wbf (f32 accum)
    dim3 g(NNODES / 64, Nout / 64);
    if (K == FIN)
      gemm_bf16_wmma<FIN><<<g, 128, 0, stream>>>((const __bf16*)xbf, (const __bf16*)wbf, bufA, Nout);
    else
      gemm_bf16_wmma<HCDIM><<<g, 128, 0, stream>>>((const __bf16*)xbf, (const __bf16*)wbf, bufA, Nout);
    // 2) per-node attention scores
    int nh = NNODES * Hh;
    att_scores<<<(nh + 255) / 256, 256, 0, stream>>>(bufA, asv, adv, ai, aj, Hh);
    // 3) clear accumulators
    fill0(mu, (size_t)nh);
    fill0(sden, (size_t)nh);
    fill0(bufB, (size_t)NNODES * Nout);
    // 4) segment softmax over incoming edges + weighted aggregation
    size_t eh = (size_t)ETOT * Hh;
    edge_max<<<(unsigned)((eh + 255) / 256), 256, 0, stream>>>(src, dst, ai, aj, mu, Hh);
    edge_p  <<<(unsigned)((eh + 255) / 256), 256, 0, stream>>>(src, dst, ai, aj, mu, sden, pbuf, Hh);
    size_t ea = eh * (CHAN / 4);
    edge_agg<<<(unsigned)((ea + 255) / 256), 256, 0, stream>>>(src, dst, bufA, pbuf, sden, bufB, Hh);
    // 5) bias + activation; intermediate layers emit bf16 for the next GEMM
    size_t tot = (size_t)NNODES * Nout;
    if (!isLast)
      bias_elu_bf<<<(unsigned)((tot + 255) / 256), 256, 0, stream>>>(bufB, bias, xbf, tot, Nout - 1);
    else
      bias_f32<<<(unsigned)((tot + 255) / 256), 256, 0, stream>>>(bufB, bias, tot, Nout - 1);
  };

  // layer-0 GEMM input: convert x once to bf16
  cvt(x, xbf, (size_t)NNODES * FIN);
  layer(FIN,   NHEADS, W0, as0, ad0, b0, 0);  // 64  -> 256, ELU (bf16 out)
  layer(HCDIM, NHEADS, W1, as1, ad1, b1, 0);  // 256 -> 256, ELU (bf16 out)
  layer(HCDIM, 1,      W2, as2, ad2, b2, 1);  // 256 -> 64, bias only (f32)

  // global max pool per graph
  fill0(pu, (size_t)NGRAPH * CHAN);
  pool_max<<<(NNODES * CHAN + 255) / 256, 256, 0, stream>>>(bufB, batch, pu);
  pool_dec<<<(NGRAPH * CHAN + 255) / 256, 256, 0, stream>>>(pu, (float*)d_out);
}
